// build_cnn4_63204738728214
// MI455X (gfx1250) — matrix operand order repaired and checked by reading, not compiled, not run on hardware
//
#include <hip/hip_runtime.h>
#include <hip/hip_bf16.h>

typedef __attribute__((ext_vector_type(16))) _Float16 v16h;
typedef __attribute__((ext_vector_type(8)))  float    v8f;

#define EPS 1e-5f

// ---------------------------------------------------------------------------
// conv1: x(32,3,512,512) --4x4 s4--> y1(32,4,128,128), +bias (pre-BN)
// ---------------------------------------------------------------------------
__global__ void conv1_k(const float* __restrict__ x, const float* __restrict__ w1,
                        const float* __restrict__ b1, float* __restrict__ y1) {
    __shared__ float ws[192];                      // w1: 4*3*4*4
    int t = threadIdx.x;
    if (t < 192) ws[t] = w1[t];
    __syncthreads();
    int pos = blockIdx.x * 256 + t;                // 32*128*128 = 524288 positions
    int n = pos >> 14, rem = pos & 16383;
    int oh = rem >> 7, ow = rem & 127;
    float acc[4];
#pragma unroll
    for (int oc = 0; oc < 4; ++oc) acc[oc] = b1[oc];
#pragma unroll
    for (int ic = 0; ic < 3; ++ic) {
        const float* base = x + (((n * 3 + ic) * 512 + oh * 4) * 512 + ow * 4);
#pragma unroll
        for (int dy = 0; dy < 4; ++dy) {
            float4 v = *(const float4*)(base + dy * 512);   // 16B-aligned b128 loads
#pragma unroll
            for (int oc = 0; oc < 4; ++oc) {
                const float* wp = ws + ((oc * 3 + ic) * 4 + dy) * 4;
                acc[oc] += v.x * wp[0] + v.y * wp[1] + v.z * wp[2] + v.w * wp[3];
            }
        }
    }
#pragma unroll
    for (int oc = 0; oc < 4; ++oc)
        y1[((n * 4 + oc) << 14) + (oh << 7) + ow] = acc[oc];
}

// ---------------------------------------------------------------------------
// Deterministic per-channel stats: stage 1 (P=64 partials per channel)
// src layout NCHW, HW = 1<<hwlog.  partials[2*(c*64+p)] = {sum, sumsq}
// ---------------------------------------------------------------------------
__global__ void stats_partial_k(const float* __restrict__ src, float* __restrict__ part,
                                int C, int Nn, int hwlog) {
    const int P = 64;
    int c = blockIdx.x / P, p = blockIdx.x % P;
    int HW = 1 << hwlog, E = Nn << hwlog;
    float s = 0.f, s2 = 0.f;
    for (int i = p * 256 + threadIdx.x; i < E; i += P * 256) {
        int n = i >> hwlog, hw = i & (HW - 1);
        float v = src[((n * C + c) << hwlog) + hw];
        s += v; s2 += v * v;
    }
    __shared__ float r1[256], r2[256];
    r1[threadIdx.x] = s; r2[threadIdx.x] = s2;
    __syncthreads();
    for (int o = 128; o > 0; o >>= 1) {
        if (threadIdx.x < o) { r1[threadIdx.x] += r1[threadIdx.x + o];
                               r2[threadIdx.x] += r2[threadIdx.x + o]; }
        __syncthreads();
    }
    if (threadIdx.x == 0) { part[2 * (c * P + p)] = r1[0]; part[2 * (c * P + p) + 1] = r2[0]; }
}

// stage 2: fold BN into per-channel scale/bias: sb[2c]=g/sqrt(v+eps), sb[2c+1]=be-m*scale
__global__ void stats_final_k(const float* __restrict__ part, int C,
                              const float* __restrict__ g, const float* __restrict__ be,
                              float Einv, float* __restrict__ sb) {
    int c = threadIdx.x;
    if (c >= C) return;
    float s = 0.f, s2 = 0.f;
    for (int p = 0; p < 64; ++p) { s += part[2 * (c * 64 + p)]; s2 += part[2 * (c * 64 + p) + 1]; }
    float m   = s * Einv;
    float var = s2 * Einv - m * m;
    float scale = g[c] * rsqrtf(var + EPS);
    sb[2 * c]     = scale;
    sb[2 * c + 1] = be[c] - m * scale;
}

// ---------------------------------------------------------------------------
// conv2: relu(bn1(y1)) --2x2 s2--> y2(32,8,64,64)
// ---------------------------------------------------------------------------
__global__ void conv2_k(const float* __restrict__ y1, const float* __restrict__ w2,
                        const float* __restrict__ b2, const float* __restrict__ sb1,
                        float* __restrict__ y2) {
    __shared__ float ws[128], sc[4], bs[4];        // w2: 8*4*2*2
    int t = threadIdx.x;
    if (t < 128) ws[t] = w2[t];
    if (t < 4) { sc[t] = sb1[2 * t]; bs[t] = sb1[2 * t + 1]; }
    __syncthreads();
    int pos = blockIdx.x * 256 + t;                // 32*64*64 = 131072 positions
    int n = pos >> 12, rem = pos & 4095;
    int oh = rem >> 6, ow = rem & 63;
    float acc[8];
#pragma unroll
    for (int oc = 0; oc < 8; ++oc) acc[oc] = b2[oc];
#pragma unroll
    for (int ic = 0; ic < 4; ++ic)
#pragma unroll
        for (int ky = 0; ky < 2; ++ky)
#pragma unroll
            for (int kx = 0; kx < 2; ++kx) {
                float v = y1[((n * 4 + ic) << 14) + ((oh * 2 + ky) << 7) + ow * 2 + kx];
                v = fmaxf(v * sc[ic] + bs[ic], 0.f);
#pragma unroll
                for (int oc = 0; oc < 8; ++oc)
                    acc[oc] += v * ws[((oc * 4 + ic) * 2 + ky) * 2 + kx];
            }
#pragma unroll
    for (int oc = 0; oc < 8; ++oc)
        y2[((n * 8 + oc) << 12) + (oh << 6) + ow] = acc[oc];
}

// ---------------------------------------------------------------------------
// conv3 via WMMA: exact 16x16x32 GEMM.
//   M = 32*32*32 output positions, N = 16 out channels, K = 8ic*2*2 = 32
// Each wave computes a 16(pos) x 16(ch) tile with one v_wmma_f32_16x16x32_f16.
// ---------------------------------------------------------------------------
__global__ void conv3_wmma_k(const float* __restrict__ y2, const float* __restrict__ w3,
                             const float* __restrict__ b3, const float* __restrict__ sb2,
                             float* __restrict__ y3) {
    __shared__ _Float16 w3s[512];                  // w3: 16*8*2*2, as f16
    __shared__ float sc[8], bs[8];
    int t = threadIdx.x;
    w3s[t]       = (_Float16)w3[t];
    w3s[t + 256] = (_Float16)w3[t + 256];
    if (t < 8) { sc[t] = sb2[2 * t]; bs[t] = sb2[2 * t + 1]; }
    __syncthreads();

    int wave = t >> 5, lane = t & 31;
    int mbase = (blockIdx.x * 8 + wave) * 16;      // 256 blocks * 8 waves * 16 = 32768
    int lanelo = lane & 15, hi = (lane >> 4) & 1;

    // A fragment: row m = mbase+lanelo; 16-bit A 16x32 layout:
    //   lanes 0-15 : halves 0-7 -> K=0..7,  halves 8-15 -> K=16..23
    //   lanes 16-31: halves 0-7 -> K=8..15, halves 8-15 -> K=24..31
    int m = mbase + lanelo;
    int nImg = m >> 10, rem = m & 1023;
    int oh = rem >> 5, ow = rem & 31;
    v16h a, b;
#pragma unroll
    for (int i = 0; i < 16; ++i) {
        int k  = (i < 8 ? i : i + 8) + hi * 8;     // GEMM K index
        int ic = k >> 2, kh = (k >> 1) & 1, kw = k & 1;
        float v = y2[((nImg * 8 + ic) << 12) + ((oh * 2 + kh) << 6) + (ow * 2 + kw)];
        v = fmaxf(v * sc[ic] + bs[ic], 0.f);       // fused bn2 + relu
        a[i] = (_Float16)v;
        // B fragment: col n = lanelo; same K layout as A (lanes 0-15: K=0..7,16..23; lanes 16-31: K=8..15,24..31)
        int kb = (i < 8 ? i : i + 8) + hi * 8;
        b[i] = w3s[lanelo * 32 + kb];              // B[k][n] = w3[n*32 + k]
    }
    v8f c = {};
    v8f d = __builtin_amdgcn_wmma_f32_16x16x32_f16(false, a, false, b,
                                                   (short)0, c, false, false);
    // D: lane -> col(channel) = lanelo; VGPR j -> row m = mbase + j + 8*hi
    int ch = lanelo;
    float bias = b3[ch];
#pragma unroll
    for (int j = 0; j < 8; ++j) {
        int mm = mbase + j + hi * 8;
        int nI = mm >> 10, r2 = mm & 1023;
        y3[((nI * 16 + ch) << 10) + r2] = d[j] + bias;
    }
}

// ---------------------------------------------------------------------------
// conv4: relu(bn3(y3)) --1x1--> y4(32,4,32,32)
// ---------------------------------------------------------------------------
__global__ void conv4_k(const float* __restrict__ y3, const float* __restrict__ w4,
                        const float* __restrict__ b4, const float* __restrict__ sb3,
                        float* __restrict__ y4) {
    __shared__ float ws[64], sc[16], bs[16];       // w4: 4*16
    int t = threadIdx.x;
    if (t < 64) ws[t] = w4[t];
    if (t < 16) { sc[t] = sb3[2 * t]; bs[t] = sb3[2 * t + 1]; }
    __syncthreads();
    int pos = blockIdx.x * 256 + t;                // 32*1024 = 32768 positions
    int n = pos >> 10, hw = pos & 1023;
    float acc[4];
#pragma unroll
    for (int oc = 0; oc < 4; ++oc) acc[oc] = b4[oc];
#pragma unroll
    for (int ic = 0; ic < 16; ++ic) {
        float v = fmaxf(y3[((n * 16 + ic) << 10) + hw] * sc[ic] + bs[ic], 0.f);
#pragma unroll
        for (int oc = 0; oc < 4; ++oc) acc[oc] += v * ws[oc * 16 + ic];
    }
#pragma unroll
    for (int oc = 0; oc < 4; ++oc)
        y4[((n * 4 + oc) << 10) + hw] = acc[oc];
}

// ---------------------------------------------------------------------------
// final: conv5 (1x1, 4->1) on relu(bn4(y4)), fused 16x16 nearest upsample
// ---------------------------------------------------------------------------
__global__ void final_k(const float* __restrict__ y4, const float* __restrict__ w5,
                        const float* __restrict__ b5, const float* __restrict__ sb4,
                        float* __restrict__ out) {
    __shared__ float sc[4], bs[4], wv[4];
    int t = threadIdx.x;
    if (t < 4) { sc[t] = sb4[2 * t]; bs[t] = sb4[2 * t + 1]; wv[t] = w5[t]; }
    __syncthreads();
    int pos = blockIdx.x * 256 + t;                // 32768 positions
    int n = pos >> 10, oh = (pos >> 5) & 31, ow = pos & 31;
    float z = b5[0];
#pragma unroll
    for (int ic = 0; ic < 4; ++ic) {
        float v = fmaxf(y4[((n * 4 + ic) << 10) + (oh << 5) + ow] * sc[ic] + bs[ic], 0.f);
        z += v * wv[ic];
    }
    float4 zz = make_float4(z, z, z, z);
#pragma unroll
    for (int r = 0; r < 16; ++r) {
        float4* p = (float4*)(out + ((long)n * 512 + (oh * 16 + r)) * 512 + ow * 16);
        p[0] = zz; p[1] = zz; p[2] = zz; p[3] = zz;   // coalesced b128 stores
    }
}

// ---------------------------------------------------------------------------
extern "C" void kernel_launch(void* const* d_in, const int* in_sizes, int n_in,
                              void* d_out, int out_size, void* d_ws, size_t ws_size,
                              hipStream_t stream) {
    const float* x   = (const float*)d_in[0];
    const float* w1  = (const float*)d_in[1];
    const float* b1  = (const float*)d_in[2];
    const float* g1  = (const float*)d_in[3];
    const float* be1 = (const float*)d_in[4];
    const float* w2  = (const float*)d_in[5];
    const float* b2  = (const float*)d_in[6];
    const float* g2  = (const float*)d_in[7];
    const float* be2 = (const float*)d_in[8];
    const float* w3  = (const float*)d_in[9];
    const float* b3  = (const float*)d_in[10];
    const float* g3  = (const float*)d_in[11];
    const float* be3 = (const float*)d_in[12];
    const float* w4  = (const float*)d_in[13];
    const float* b4  = (const float*)d_in[14];
    const float* g4  = (const float*)d_in[15];
    const float* be4 = (const float*)d_in[16];
    const float* w5  = (const float*)d_in[17];
    const float* b5  = (const float*)d_in[18];
    float* out = (float*)d_out;

    // workspace layout (floats), ~14.6 MB total — resident in L2
    float* W    = (float*)d_ws;
    float* y1   = W;                  // 32*4*128*128 = 2097152
    float* y2   = y1 + 2097152;       // 32*8*64*64   = 1048576
    float* y3   = y2 + 1048576;       // 32*16*32*32  = 524288
    float* y4   = y3 + 524288;        // 32*4*32*32   = 131072
    float* part = y4 + 131072;        // 16ch * 64p * 2 = 2048
    float* sb1  = part + 2048;        // 8
    float* sb2  = sb1 + 8;            // 16
    float* sb3  = sb2 + 16;           // 32
    float* sb4  = sb3 + 32;           // 8

    dim3 blk(256);

    conv1_k<<<2048, blk, 0, stream>>>(x, w1, b1, y1);
    stats_partial_k<<<4 * 64, blk, 0, stream>>>(y1, part, 4, 32, 14);
    stats_final_k<<<1, 32, 0, stream>>>(part, 4, g1, be1, 1.f / 524288.f, sb1);

    conv2_k<<<512, blk, 0, stream>>>(y1, w2, b2, sb1, y2);
    stats_partial_k<<<8 * 64, blk, 0, stream>>>(y2, part, 8, 32, 12);
    stats_final_k<<<1, 32, 0, stream>>>(part, 8, g2, be2, 1.f / 131072.f, sb2);

    conv3_wmma_k<<<256, blk, 0, stream>>>(y2, w3, b3, sb2, y3);
    stats_partial_k<<<16 * 64, blk, 0, stream>>>(y3, part, 16, 32, 10);
    stats_final_k<<<1, 32, 0, stream>>>(part, 16, g3, be3, 1.f / 32768.f, sb3);

    conv4_k<<<128, blk, 0, stream>>>(y3, w4, b4, sb3, y4);
    stats_partial_k<<<4 * 64, blk, 0, stream>>>(y4, part, 4, 32, 10);
    stats_final_k<<<1, 32, 0, stream>>>(part, 4, g4, be4, 1.f / 32768.f, sb4);

    final_k<<<128, blk, 0, stream>>>(y4, w5, b5, sb4, out);
}
